// _AttnBlock1D_49606872268944
// MI455X (gfx1250) — compile-verified, matrix operand order checked by reading, not run on hardware
//
#include <hip/hip_runtime.h>
#include <hip/hip_bf16.h>
#include <stddef.h>

// ---------------------------------------------------------------------------
// AttnBlock1D for MI455X (gfx1250, wave32, WMMA bf16 16x16x32, f32 acc)
// B=8, C=512, L=4096, G=32 groups.
// ---------------------------------------------------------------------------

typedef __attribute__((ext_vector_type(16))) __bf16 v16bf;
typedef __attribute__((ext_vector_type(8)))  __bf16 v8bf;
typedef __attribute__((ext_vector_type(8)))  float  v8f;

static constexpr int BB = 8;
static constexpr int CC = 512;
static constexpr int LL = 4096;
static constexpr int JB = 1024;   // attention j-chunk held in LDS

__device__ __forceinline__ v8f vzero8() {
  v8f z;
#pragma unroll
  for (int i = 0; i < 8; ++i) z[i] = 0.0f;
  return z;
}

// Load one 16x32 bf16 fragment (A layout, or B^T stored row-major) per the
// CDNA5 ISA 16-bit matrix VGPR striping:
//   lane<16 : row=lane,    K runs [0..7]  and [16..23]
//   lane>=16: row=lane-16, K runs [8..15] and [24..31]
// Two 16-byte loads per lane; base must be 16B aligned (all our strides are).
__device__ __forceinline__ v16bf load_frag(const __bf16* base, int stride,
                                           int lr, int half) {
  const __bf16* p = base + (size_t)lr * stride + half * 8;
  v8bf lo = *(const v8bf*)(p);
  v8bf hi = *(const v8bf*)(p + 16);
  return __builtin_shufflevector(lo, hi, 0, 1, 2, 3, 4, 5, 6, 7,
                                 8, 9, 10, 11, 12, 13, 14, 15);
}

__device__ __forceinline__ v8f wmma_bf16(v16bf a, v16bf b, v8f c) {
  return __builtin_amdgcn_wmma_f32_16x16x32_bf16(
      /*neg_a=*/false, a, /*neg_b=*/false, b,
      /*c_mod=*/(short)0, c, /*reuse_a=*/false, /*reuse_b=*/false);
}

// ---------------------------------------------------------------------------
// K0: convert the four f32 weight matrices to bf16 (row-major [o][c]).
// ---------------------------------------------------------------------------
__global__ void cvt_weights(const float* __restrict__ a, const float* __restrict__ b,
                            const float* __restrict__ c, const float* __restrict__ d,
                            __bf16* __restrict__ oa, __bf16* __restrict__ ob,
                            __bf16* __restrict__ oc, __bf16* __restrict__ od, int n) {
  int i = blockIdx.x * blockDim.x + threadIdx.x;
  if (i < n) {
    oa[i] = (__bf16)a[i];
    ob[i] = (__bf16)b[i];
    oc[i] = (__bf16)c[i];
    od[i] = (__bf16)d[i];
  }
}

// ---------------------------------------------------------------------------
// K1: group-norm statistics. One 256-thread block per (b, g); reduce over
// 16 channels x 4096 = 65536 f32. stats[bg] = {mean, rstd}.
// ---------------------------------------------------------------------------
__global__ void gn_stats(const float* __restrict__ x, float* __restrict__ stats) {
  const int bg = blockIdx.x;
  const int b = bg >> 5;
  const int g = bg & 31;
  const float* base = x + ((size_t)b * CC + g * 16) * LL;
  float s = 0.f, q = 0.f;
  for (int e = threadIdx.x; e < 16 * LL; e += 256) {
    float v = base[e];
    s += v;
    q += v * v;
  }
  __shared__ float ss[256], sq[256];
  ss[threadIdx.x] = s;
  sq[threadIdx.x] = q;
  __syncthreads();
  for (int st = 128; st > 0; st >>= 1) {
    if (threadIdx.x < st) {
      ss[threadIdx.x] += ss[threadIdx.x + st];
      sq[threadIdx.x] += sq[threadIdx.x + st];
    }
    __syncthreads();
  }
  if (threadIdx.x == 0) {
    const float n = 16.0f * (float)LL;
    float mean = ss[0] / n;
    float var = sq[0] / n - mean * mean;
    stats[bg * 2 + 0] = mean;
    stats[bg * 2 + 1] = rsqrtf(var + 1e-6f);
  }
}

// ---------------------------------------------------------------------------
// K2: apply group-norm and write h transposed as bf16 [b][l][c]
// (the layout every downstream GEMM wants for its B^T fragments).
// Block (32,8) moves a 32l x 32c tile through LDS.
// ---------------------------------------------------------------------------
__global__ void gn_apply_t(const float* __restrict__ x, const float* __restrict__ gw,
                           const float* __restrict__ gb, const float* __restrict__ stats,
                           __bf16* __restrict__ ht) {
  __shared__ float tile[32][33];
  const int b = blockIdx.z;
  const int lt = blockIdx.x * 32;
  const int c0 = blockIdx.y * 32;
  const int tx = threadIdx.x;  // 0..31 (l within tile on read, c on write)
  const int ty = threadIdx.y;  // 0..7
  for (int cc = ty; cc < 32; cc += 8) {
    const int c = c0 + cc;
    const int g = c >> 4;
    const float mean = stats[(b * 32 + g) * 2 + 0];
    const float rstd = stats[(b * 32 + g) * 2 + 1];
    float v = x[((size_t)b * CC + c) * LL + lt + tx];
    tile[cc][tx] = (v - mean) * rstd * gw[c] + gb[c];
  }
  __syncthreads();
  for (int rr = ty; rr < 32; rr += 8) {
    ht[((size_t)b * LL + lt + rr) * CC + c0 + tx] = (__bf16)tile[tx][rr];
  }
}

// ---------------------------------------------------------------------------
// K3: 1x1 conv GEMM  out[o][l] = sum_c W[o][c] * h[c][l] + bias[o].
// A = W (row-major), B^T = h_t[b][l][c]. 4 waves/block, 32x64 tile per wave
// (acc[2][4] = 64 VGPRs keeps the accumulators out of scratch).
// mode 0: store transposed bf16 [b][l][o] (for Q and K; packed 16B stores)
// mode 1: store natural    bf16 [b][o][l] (for V)
// ---------------------------------------------------------------------------
__global__ void __launch_bounds__(128)
    __attribute__((amdgpu_waves_per_eu(1)))
    gemm_qkv(const __bf16* __restrict__ W, const float* __restrict__ bias,
             const __bf16* __restrict__ ht, __bf16* __restrict__ out, int mode) {
  const int b = blockIdx.z;
  const int M0 = blockIdx.x * 32;
  const int wave = threadIdx.x >> 5;
  const int lane = threadIdx.x & 31;
  const int lr = lane & 15;
  const int half = lane >> 4;
  const int N0 = blockIdx.y * 256 + wave * 64;

  v8f acc[2][4];
#pragma unroll
  for (int mi = 0; mi < 2; ++mi)
#pragma unroll
    for (int ni = 0; ni < 4; ++ni) acc[mi][ni] = vzero8();

  const __bf16* hb = ht + ((size_t)b * LL + N0) * CC;
  for (int k0 = 0; k0 < CC; k0 += 32) {
    v16bf bfr[4];
#pragma unroll
    for (int ni = 0; ni < 4; ++ni)
      bfr[ni] = load_frag(hb + (size_t)(16 * ni) * CC + k0, CC, lr, half);
#pragma unroll
    for (int mi = 0; mi < 2; ++mi) {
      v16bf afr = load_frag(W + (size_t)(M0 + 16 * mi) * CC + k0, CC, lr, half);
#pragma unroll
      for (int ni = 0; ni < 4; ++ni)
        acc[mi][ni] = wmma_bf16(afr, bfr[ni], acc[mi][ni]);
    }
  }

#pragma unroll
  for (int mi = 0; mi < 2; ++mi) {
    float bv[8];
#pragma unroll
    for (int r = 0; r < 8; ++r) bv[r] = bias[M0 + 16 * mi + r + 8 * half];
#pragma unroll
    for (int ni = 0; ni < 4; ++ni) {
      if (mode == 0) {
        const int l = N0 + 16 * ni + lr;
        __bf16* dst = out + ((size_t)b * LL + l) * CC + M0 + 16 * mi + 8 * half;
        v8bf pk;
#pragma unroll
        for (int r = 0; r < 8; ++r) pk[r] = (__bf16)(acc[mi][ni][r] + bv[r]);
        *(v8bf*)dst = pk;
      } else {
#pragma unroll
        for (int r = 0; r < 8; ++r) {
          out[((size_t)b * CC + M0 + 16 * mi + r + 8 * half) * LL + N0 + 16 * ni + lr] =
              (__bf16)(acc[mi][ni][r] + bv[r]);
        }
      }
    }
  }
}

// ---------------------------------------------------------------------------
// K4: flash attention. One workgroup (8 waves) per (b, 16-row i-tile).
// Per j-chunk of 1024:
//   A: 8 waves each compute 8 S tiles (16x16, K=512 via 16 WMMAs) -> LDS f32
//   B: online-softmax stats per row; P = exp(S - m) -> LDS bf16; alpha,l in LDS
//   C: each wave owns a 64-wide c slice: rescale acc by alpha, then PV WMMAs
//      (A-frag = P from LDS, B-frag = V [b][c][l] from global/L2)
// Epilogue: divide by l, store h_att as bf16 [b][i][c].
// Q/K fragments are reloaded per step from L2 (all of Q/K/V ~100MB fits the
// 192MB L2); keeping them resident caused VGPR spills which are far worse.
// ---------------------------------------------------------------------------
__global__ void __launch_bounds__(256)
    __attribute__((amdgpu_waves_per_eu(1)))
    attn_flash(const __bf16* __restrict__ qt, const __bf16* __restrict__ kt,
               const __bf16* __restrict__ vm, __bf16* __restrict__ hout) {
  extern __shared__ char smem[];
  float* Sbuf = (float*)smem;                              // [16][JB] f32
  __bf16* Pbuf = (__bf16*)(smem + (size_t)16 * JB * 4);    // [16][JB] bf16
  float* rowm = (float*)(smem + (size_t)16 * JB * 6);      // [16]
  float* rowl = rowm + 16;                                 // [16]
  float* rowa = rowl + 16;                                 // [16]

  const int b = blockIdx.y;
  const int it = blockIdx.x * 16;
  const int tid = threadIdx.x;
  const int wave = tid >> 5;
  const int lane = tid & 31;
  const int lr = lane & 15;
  const int half = lane >> 4;
  const int ct = wave * 64;  // this wave's 64-wide c slice of the PV output
  const float scale = 0.04419417382415922f;  // 512^-0.5

  if (tid < 16) { rowm[tid] = -3.0e38f; rowl[tid] = 0.0f; }

  const __bf16* qbase = qt + ((size_t)b * LL + it) * CC;

  v8f oacc[4];
#pragma unroll
  for (int ni = 0; ni < 4; ++ni) oacc[ni] = vzero8();

  __syncthreads();

  for (int chunk = 0; chunk < LL / JB; ++chunk) {
    const int j0 = chunk * JB;

    // ---- Phase A: S tiles for this chunk (wave handles 8 of 64 tiles) ----
    for (int t = 0; t < 8; ++t) {
      const int jloc = (wave * 8 + t) * 16;
      const __bf16* kbase = kt + ((size_t)b * LL + j0 + jloc) * CC;
      v8f sa = vzero8();
#pragma unroll
      for (int kk = 0; kk < 16; ++kk) {
        v16bf qf = load_frag(qbase + kk * 32, CC, lr, half);
        v16bf kf = load_frag(kbase + kk * 32, CC, lr, half);
        sa = wmma_bf16(qf, kf, sa);
      }
#pragma unroll
      for (int r = 0; r < 8; ++r)
        Sbuf[(size_t)(r + 8 * half) * JB + jloc + lr] = sa[r] * scale;
    }
    __syncthreads();

    // ---- Phase B: online softmax stats; write P (bf16) ----
    {
      const int row = 2 * wave + half;  // each 16-lane half owns one row
      float mloc = -3.0e38f;
      for (int k = lr; k < JB; k += 16) mloc = fmaxf(mloc, Sbuf[(size_t)row * JB + k]);
#pragma unroll
      for (int m = 1; m < 16; m <<= 1) mloc = fmaxf(mloc, __shfl_xor(mloc, m));
      const float mold = rowm[row];
      const float mnew = fmaxf(mold, mloc);
      const float alpha = __expf(mold - mnew);
      float sloc = 0.0f;
      for (int k = lr; k < JB; k += 16) {
        float e = __expf(Sbuf[(size_t)row * JB + k] - mnew);
        Pbuf[(size_t)row * JB + k] = (__bf16)e;
        sloc += e;
      }
#pragma unroll
      for (int m = 1; m < 16; m <<= 1) sloc += __shfl_xor(sloc, m);
      if (lr == 0) {
        rowm[row] = mnew;
        rowl[row] = alpha * rowl[row] + sloc;
        rowa[row] = alpha;
      }
    }
    __syncthreads();

    // ---- Phase C: rescale accumulators, then PV for this chunk ----
    {
      float a8[8];
#pragma unroll
      for (int r = 0; r < 8; ++r) a8[r] = rowa[r + 8 * half];
#pragma unroll
      for (int ni = 0; ni < 4; ++ni)
#pragma unroll
        for (int r = 0; r < 8; ++r) oacc[ni][r] *= a8[r];

      for (int ks = 0; ks < JB / 32; ++ks) {
        v16bf pa = load_frag(Pbuf + ks * 32, JB, lr, half);
#pragma unroll
        for (int ni = 0; ni < 4; ++ni) {
          const __bf16* vbase =
              vm + ((size_t)b * CC + ct + 16 * ni) * LL + j0 + ks * 32;
          v16bf vb = load_frag(vbase, LL, lr, half);
          oacc[ni] = wmma_bf16(pa, vb, oacc[ni]);
        }
      }
    }
    // (no barrier needed: next-iteration barriers order S/P reuse correctly)
  }

  // ---- Epilogue: normalize and store h_att as bf16 [b][i][c] ----
  float linv[8];
#pragma unroll
  for (int r = 0; r < 8; ++r) linv[r] = 1.0f / rowl[r + 8 * half];
  __bf16* obase = hout + ((size_t)b * LL + it) * CC + ct;
#pragma unroll
  for (int ni = 0; ni < 4; ++ni)
#pragma unroll
    for (int r = 0; r < 8; ++r)
      obase[(size_t)(r + 8 * half) * CC + 16 * ni + lr] =
          (__bf16)(oacc[ni][r] * linv[r]);
}

// ---------------------------------------------------------------------------
// K5: output projection + residual: out = x + Wo * h_att + bo  (f32 out).
// Same 32x64 tiling as gemm_qkv; B^T = h_att_t[b][l][c'].
// ---------------------------------------------------------------------------
__global__ void __launch_bounds__(128)
    __attribute__((amdgpu_waves_per_eu(1)))
    gemm_out(const __bf16* __restrict__ Wo, const float* __restrict__ bo,
             const __bf16* __restrict__ hat, const float* __restrict__ x,
             float* __restrict__ out) {
  const int b = blockIdx.z;
  const int M0 = blockIdx.x * 32;
  const int wave = threadIdx.x >> 5;
  const int lane = threadIdx.x & 31;
  const int lr = lane & 15;
  const int half = lane >> 4;
  const int N0 = blockIdx.y * 256 + wave * 64;

  v8f acc[2][4];
#pragma unroll
  for (int mi = 0; mi < 2; ++mi)
#pragma unroll
    for (int ni = 0; ni < 4; ++ni) acc[mi][ni] = vzero8();

  const __bf16* hb = hat + ((size_t)b * LL + N0) * CC;
  for (int k0 = 0; k0 < CC; k0 += 32) {
    v16bf bfr[4];
#pragma unroll
    for (int ni = 0; ni < 4; ++ni)
      bfr[ni] = load_frag(hb + (size_t)(16 * ni) * CC + k0, CC, lr, half);
#pragma unroll
    for (int mi = 0; mi < 2; ++mi) {
      v16bf afr = load_frag(Wo + (size_t)(M0 + 16 * mi) * CC + k0, CC, lr, half);
#pragma unroll
      for (int ni = 0; ni < 4; ++ni)
        acc[mi][ni] = wmma_bf16(afr, bfr[ni], acc[mi][ni]);
    }
  }

#pragma unroll
  for (int mi = 0; mi < 2; ++mi) {
    float bv[8];
#pragma unroll
    for (int r = 0; r < 8; ++r) bv[r] = bo[M0 + 16 * mi + r + 8 * half];
#pragma unroll
    for (int ni = 0; ni < 4; ++ni) {
#pragma unroll
      for (int r = 0; r < 8; ++r) {
        size_t idx = ((size_t)b * CC + M0 + 16 * mi + r + 8 * half) * LL +
                     N0 + 16 * ni + lr;
        out[idx] = x[idx] + acc[mi][ni][r] + bv[r];
      }
    }
  }
}

// ---------------------------------------------------------------------------
// Host launcher
// ---------------------------------------------------------------------------
extern "C" void kernel_launch(void* const* d_in, const int* in_sizes, int n_in,
                              void* d_out, int out_size, void* d_ws, size_t ws_size,
                              hipStream_t stream) {
  (void)in_sizes; (void)n_in; (void)out_size; (void)ws_size;
  const float* x   = (const float*)d_in[0];
  const float* gnw = (const float*)d_in[1];
  const float* gnb = (const float*)d_in[2];
  const float* wq  = (const float*)d_in[3];
  const float* bq  = (const float*)d_in[4];
  const float* wk  = (const float*)d_in[5];
  const float* bk  = (const float*)d_in[6];
  const float* wv  = (const float*)d_in[7];
  const float* bv  = (const float*)d_in[8];
  const float* wo  = (const float*)d_in[9];
  const float* bo  = (const float*)d_in[10];
  float* out = (float*)d_out;

  // Workspace layout
  char* ws = (char*)d_ws;
  size_t off = 0;
  float* stats = (float*)(ws + off); off += 4096;                 // [B*G][2]
  const size_t WSZ = (size_t)CC * CC * sizeof(__bf16);
  __bf16* wqb = (__bf16*)(ws + off); off += WSZ;
  __bf16* wkb = (__bf16*)(ws + off); off += WSZ;
  __bf16* wvb = (__bf16*)(ws + off); off += WSZ;
  __bf16* wob = (__bf16*)(ws + off); off += WSZ;
  const size_t HSZ = (size_t)BB * LL * CC * sizeof(__bf16);       // 33.5 MB
  __bf16* h_t = (__bf16*)(ws + off); off += HSZ;   // [b][l][c] groupnorm out
  __bf16* q_t = (__bf16*)(ws + off); off += HSZ;   // [b][l][c]
  __bf16* k_t = (__bf16*)(ws + off); off += HSZ;   // [b][l][c]
  __bf16* v_m = (__bf16*)(ws + off); off += HSZ;   // [b][c][l]
  __bf16* hat = h_t;  // reuse: h_t dead after QKV projections

  cvt_weights<<<(CC * CC + 255) / 256, 256, 0, stream>>>(wq, wk, wv, wo,
                                                         wqb, wkb, wvb, wob, CC * CC);
  gn_stats<<<BB * 32, 256, 0, stream>>>(x, stats);
  gn_apply_t<<<dim3(LL / 32, CC / 32, BB), dim3(32, 8), 0, stream>>>(
      x, gnw, gnb, stats, h_t);

  gemm_qkv<<<dim3(CC / 32, LL / 256, BB), 128, 0, stream>>>(wqb, bq, h_t, q_t, 0);
  gemm_qkv<<<dim3(CC / 32, LL / 256, BB), 128, 0, stream>>>(wkb, bk, h_t, k_t, 0);
  gemm_qkv<<<dim3(CC / 32, LL / 256, BB), 128, 0, stream>>>(wvb, bv, h_t, v_m, 1);

  const size_t smem = (size_t)16 * JB * 6 + 256;  // S f32 + P bf16 + stats
  attn_flash<<<dim3(LL / 16, BB), 256, smem, stream>>>(q_t, k_t, v_m, hat);

  gemm_out<<<dim3(CC / 32, LL / 256, BB), 128, 0, stream>>>(wob, bo, hat, x, out);
}
